// MultiHeadAttention_33260226740516
// MI455X (gfx1250) — matrix operand order repaired and checked by reading, not compiled, not run on hardware
//
#include <hip/hip_runtime.h>

// ---------------------------------------------------------------------------
// MHA block for MI455X (gfx1250, wave32).  All GEMMs use V_WMMA_F32_16X16X32_BF16
// (fp32 inputs converted to bf16 RNE during LDS staging; fp32 accumulation).
// GEMM: 128x64 block tile, 32x64 wave tile, double-buffered LDS, 1 barrier/K-step.
// Requires >= 64 MB workspace (4 regions of B*S*D floats).
// ---------------------------------------------------------------------------

#define HIDDEN 1024
#define HEADS  16
#define HEAD_W 64
#define BATCH  4
#define SEQ    1024
#define ROWS   (BATCH * SEQ)          // 4096
#define SCALE  0.03125f               // rsqrt(1024)
#define LN_EPS 1e-5f

typedef __attribute__((ext_vector_type(16))) __bf16 v16bf;
typedef __attribute__((ext_vector_type(8)))  float  v8f;

struct Frag { unsigned u[8]; };       // 32 bytes == v16bf

__device__ inline v8f wmma_bf16(const Frag& a, const Frag& b, v8f c) {
    v16bf av = __builtin_bit_cast(v16bf, a);
    v16bf bv = __builtin_bit_cast(v16bf, b);
    // (neg_a, A, neg_b, B, c_mod, C, reuse_a, reuse_b)
    return __builtin_amdgcn_wmma_f32_16x16x32_bf16(false, av, false, bv,
                                                   (short)0, c, false, false);
}

__device__ inline unsigned short f2bf(float f) {
    unsigned u = __builtin_bit_cast(unsigned, f);
    return (unsigned short)((u + 0x7FFFu + ((u >> 16) & 1u)) >> 16);  // RNE
}
__device__ inline unsigned pack2bf(float lo, float hi) {
    return (unsigned)f2bf(lo) | ((unsigned)f2bf(hi) << 16);
}

// A-matrix 16x32 bf16 fragment from LDS (ISA 7.12.2 16-bit A layout):
// lane m = lane&15 ; lanes<16 hold K={0..7,16..23}, lanes>=16 K={8..15,24..31}
__device__ inline void load_a_bf(const unsigned short* s, int stride, int rowBase,
                                 int kcol, int lane, unsigned* u) {
    int m  = lane & 15;
    int kb = (lane < 16) ? 0 : 8;
    const unsigned short* p = s + (size_t)(rowBase + m) * stride + kcol + kb;
#pragma unroll
    for (int j = 0; j < 4; ++j) {
        u[j]     = *(const unsigned*)(p + 2 * j);
        u[4 + j] = *(const unsigned*)(p + 16 + 2 * j);
    }
}

// Same A fragment but source rows are fp32 in LDS (convert+pack on load).
__device__ inline void load_a_f32(const float* s, int stride, int kcol,
                                  int lane, unsigned* u) {
    int m  = lane & 15;
    int kb = (lane < 16) ? 0 : 8;
    const float* p = s + (size_t)m * stride + kcol + kb;
#pragma unroll
    for (int j = 0; j < 4; ++j) {
        u[j]     = pack2bf(p[2 * j],      p[2 * j + 1]);
        u[4 + j] = pack2bf(p[16 + 2 * j], p[16 + 2 * j + 1]);
    }
}

// B-matrix 32x16 bf16 fragment; LDS holds rows n (the N index) with K contiguous.
// lane n = lane&15 ; lanes<16 K={0..7,16..23}, lanes>=16 K={8..15,24..31} (2 K per VGPR).
__device__ inline void load_b_bf(const unsigned short* s, int stride, int nBase,
                                 int kcol, int lane, unsigned* u) {
    int n  = lane & 15;
    int kb = (lane < 16) ? 0 : 8;
    const unsigned short* p = s + (size_t)(nBase + n) * stride + kcol + kb;
#pragma unroll
    for (int j = 0; j < 4; ++j) {
        u[j]     = *(const unsigned*)(p + 2 * j);
        u[4 + j] = *(const unsigned*)(p + 16 + 2 * j);
    }
}

// ---------------------------------------------------------------------------
// GEMM: Y[M,N] = X[M,K] @ W[N,K]^T (+bias[N]) (+residual[M,N])
// block 128 thr = 4 waves; block tile 128x64; wave tile 32x64; K-step 32.
// Double-buffered LDS; global loads for tile k+1 issued before WMMAs on tile k.
// ---------------------------------------------------------------------------
#define XSTR 34
#define WSTR 34

struct StageRegs { float4 xf[8]; float4 wf[4]; };

__device__ inline void stage_load(const float* __restrict__ X,
                                  const float* __restrict__ W, int K,
                                  int tileM, int tileN, int k0, int tid,
                                  StageRegs& r) {
    int rr = tid >> 3, cc = (tid & 7) * 4;
#pragma unroll
    for (int p = 0; p < 8; ++p)        // 128 rows of X, 32 k
        r.xf[p] = *(const float4*)(X + (size_t)(tileM + p * 16 + rr) * K + k0 + cc);
#pragma unroll
    for (int p = 0; p < 4; ++p)        // 64 rows of W, 32 k
        r.wf[p] = *(const float4*)(W + (size_t)(tileN + p * 16 + rr) * K + k0 + cc);
}

__device__ inline void stage_store(unsigned short* sX, unsigned short* sW,
                                   int tid, const StageRegs& r) {
    int rr = tid >> 3, cc = (tid & 7) * 4;
#pragma unroll
    for (int p = 0; p < 8; ++p) {
        unsigned* d = (unsigned*)(sX + (p * 16 + rr) * XSTR + cc);
        d[0] = pack2bf(r.xf[p].x, r.xf[p].y);
        d[1] = pack2bf(r.xf[p].z, r.xf[p].w);
    }
#pragma unroll
    for (int p = 0; p < 4; ++p) {
        unsigned* d = (unsigned*)(sW + (p * 16 + rr) * WSTR + cc);
        d[0] = pack2bf(r.wf[p].x, r.wf[p].y);
        d[1] = pack2bf(r.wf[p].z, r.wf[p].w);
    }
}

__global__ __launch_bounds__(128) void gemm_xwt_bf16(
    const float* __restrict__ X, const float* __restrict__ W,
    const float* __restrict__ bias, const float* __restrict__ residual,
    float* __restrict__ Y, int M, int N, int K)
{
    __shared__ unsigned short sX[2][128 * XSTR];
    __shared__ unsigned short sW[2][64 * WSTR];

    const int tid   = threadIdx.x;
    const int lane  = tid & 31;
    const int wave  = tid >> 5;        // 0..3 -> rows wave*32 .. +31
    const int tileN = blockIdx.x * 64;
    const int tileM = blockIdx.y * 128;

    v8f acc[2][4] = {};

    StageRegs r;
    stage_load(X, W, K, tileM, tileN, 0, tid, r);
    stage_store(sX[0], sW[0], tid, r);
    __syncthreads();

    const int nk = K / 32;
    for (int it = 0; it < nk; ++it) {
        const int cur = it & 1;
        const bool hasNext = (it + 1) < nk;

        // issue next tile's global loads first (overlap with WMMA below)
        if (hasNext) stage_load(X, W, K, tileM, tileN, (it + 1) * 32, tid, r);

        // matrix math on current LDS buffer
        Frag a0, a1;
        load_a_bf(sX[cur], XSTR, wave * 32,      0, lane, a0.u);
        load_a_bf(sX[cur], XSTR, wave * 32 + 16, 0, lane, a1.u);
#pragma unroll
        for (int ns = 0; ns < 4; ++ns) {
            Frag bfr; load_b_bf(sW[cur], WSTR, ns * 16, 0, lane, bfr.u);
            acc[0][ns] = wmma_bf16(a0, bfr, acc[0][ns]);
            acc[1][ns] = wmma_bf16(a1, bfr, acc[1][ns]);
        }

        // convert + park next tile into the other buffer
        if (hasNext) stage_store(sX[cur ^ 1], sW[cur ^ 1], tid, r);
        __syncthreads();
    }

    // C/D layout: vgpr r, lanes<16 -> M=r, lanes>=16 -> M=8+r, N=lane&15
    int mh = lane >> 4, nl = lane & 15;
#pragma unroll
    for (int sub = 0; sub < 2; ++sub) {
#pragma unroll
        for (int ns = 0; ns < 4; ++ns) {
            int col = tileN + ns * 16 + nl;
            float bv = bias ? bias[col] : 0.0f;
#pragma unroll
            for (int rr = 0; rr < 8; ++rr) {
                int row = tileM + wave * 32 + sub * 16 + mh * 8 + rr;
                size_t idx = (size_t)row * N + col;
                float vv = acc[sub][ns][rr] + bv;
                if (residual) vv += residual[idx];
                Y[idx] = vv;
            }
        }
    }
}

// ---------------------------------------------------------------------------
// Fused attention: per (head*BATCH+batch, 16-query tile)
//   scores = scale * Q Kt  -> LDS ; softmax ; write attn out ; ctx = P @ V
// block 256 thr = 8 waves.
// ---------------------------------------------------------------------------
#define QSTR 66
#define KSTR 66
#define SSTR 1032
#define VSTR 130

__global__ __launch_bounds__(256) void attn_fused(
    const float* __restrict__ Qh, const float* __restrict__ Kh,
    const float* __restrict__ Vh, const unsigned char* __restrict__ mask,
    float* __restrict__ attn_out, float* __restrict__ Ctx)
{
    __shared__ unsigned short sQ[16 * QSTR];
    __shared__ unsigned short sK[128 * KSTR];
    __shared__ float          sS[16 * SSTR];
    __shared__ unsigned short sVt[2][64 * VSTR];
    __shared__ float          sOut[8 * 256];

    const int tid  = threadIdx.x;
    const int lane = tid & 31;
    const int wave = tid >> 5;
    const int qt   = blockIdx.x;          // 0..63
    const int hb   = blockIdx.y;          // n = h*BATCH + b
    const int h    = hb >> 2;
    const int b    = hb & 3;
    const int mh   = lane >> 4, nl = lane & 15;

    // ---- stage Q tile (16 x 64) as bf16 -------------------------------
    if (tid < 128) {
        int row = tid >> 3, kp = (tid & 7) * 8;
        const float* g = Qh + ((size_t)b * SEQ + qt * 16 + row) * HIDDEN + h * HEAD_W + kp;
        float4 f0 = *(const float4*)(g);
        float4 f1 = *(const float4*)(g + 4);
        unsigned* d = (unsigned*)(sQ + row * QSTR + kp);
        d[0] = pack2bf(f0.x, f0.y); d[1] = pack2bf(f0.z, f0.w);
        d[2] = pack2bf(f1.x, f1.y); d[3] = pack2bf(f1.z, f1.w);
    }
    __syncthreads();

    Frag aQ0, aQ1;
    load_a_bf(sQ, QSTR, 0, 0,  lane, aQ0.u);
    load_a_bf(sQ, QSTR, 0, 32, lane, aQ1.u);

    // ---- phase 1: scores = scale * Q K^T (+mask) ----------------------
    for (int kb = 0; kb < 8; ++kb) {      // 128 keys per iteration
        {   // stage K tile: 128 keys x 64 dims
            int row = tid >> 1, kp = (tid & 1) * 32;
            const float* g = Kh + ((size_t)b * SEQ + kb * 128 + row) * HIDDEN + h * HEAD_W + kp;
            unsigned* d = (unsigned*)(sK + row * KSTR + kp);
#pragma unroll
            for (int i = 0; i < 8; ++i) {
                float4 f = *(const float4*)(g + 4 * i);
                d[2 * i]     = pack2bf(f.x, f.y);
                d[2 * i + 1] = pack2bf(f.z, f.w);
            }
        }
        __syncthreads();
        Frag b0, b1;
        load_b_bf(sK, KSTR, wave * 16, 0,  lane, b0.u);
        load_b_bf(sK, KSTR, wave * 16, 32, lane, b1.u);
        v8f s = {};
        s = wmma_bf16(aQ0, b0, s);
        s = wmma_bf16(aQ1, b1, s);
#pragma unroll
        for (int r = 0; r < 8; ++r) {
            int qr  = mh * 8 + r;
            int key = kb * 128 + wave * 16 + nl;
            unsigned char mv = mask[(size_t)b * SEQ * SEQ + (size_t)(qt * 16 + qr) * SEQ + key];
            sS[qr * SSTR + key] = mv ? -3.0e38f : s[r] * SCALE;
        }
        __syncthreads();
    }

    // ---- softmax (one wave per 2 rows), write normalized attn ---------
#pragma unroll
    for (int rr = 0; rr < 2; ++rr) {
        int row = wave * 2 + rr;
        float* sr = sS + row * SSTR;
        float m = -3.4e38f;
        for (int i = 0; i < 32; ++i) m = fmaxf(m, sr[lane + 32 * i]);
        for (int off = 16; off; off >>= 1) m = fmaxf(m, __shfl_xor(m, off));
        float sum = 0.f;
        for (int i = 0; i < 32; ++i) {
            int c = lane + 32 * i;
            float p = __expf(sr[c] - m);
            sr[c] = p;
            sum += p;
        }
        for (int off = 16; off; off >>= 1) sum += __shfl_xor(sum, off);
        float inv = 1.0f / sum;
        float* ag = attn_out + (size_t)hb * SEQ * SEQ + (size_t)(qt * 16 + row) * SEQ;
        for (int i = 0; i < 32; ++i) {
            int c = lane + 32 * i;
            float pn = sr[c] * inv;
            sr[c] = pn;
            ag[c] = pn;
        }
    }
    __syncthreads();

    // ---- phase 2: ctx = P @ V  (wave -> nsub = wave&3, khalf = wave>>2) ----
    const int nsub  = wave & 3;
    const int khalf = wave >> 2;
    v8f o = {};
    for (int kb2 = 0; kb2 < 4; ++kb2) {
        {   // stage V transposed (dim-major) for both K-halves
            int key = tid >> 1, dp = (tid & 1) * 32;
#pragma unroll
            for (int blk = 0; blk < 2; ++blk) {
                const float* g = Vh + ((size_t)b * SEQ + blk * 512 + kb2 * 128 + key) * HIDDEN
                                 + h * HEAD_W + dp;
#pragma unroll
                for (int i = 0; i < 32; i += 4) {
                    float4 f = *(const float4*)(g + i);
                    sVt[blk][(dp + i + 0) * VSTR + key] = f2bf(f.x);
                    sVt[blk][(dp + i + 1) * VSTR + key] = f2bf(f.y);
                    sVt[blk][(dp + i + 2) * VSTR + key] = f2bf(f.z);
                    sVt[blk][(dp + i + 3) * VSTR + key] = f2bf(f.w);
                }
            }
        }
        __syncthreads();
#pragma unroll
        for (int ks = 0; ks < 4; ++ks) {
            Frag a; load_a_f32(sS, SSTR, khalf * 512 + kb2 * 128 + ks * 32, lane, a.u);
            Frag v; load_b_bf(sVt[khalf], VSTR, nsub * 16, ks * 32, lane, v.u);
            o = wmma_bf16(a, v, o);
        }
        __syncthreads();
    }

    // partial store + cross-wave reduction (pairs w and w+4 share nsub)
#pragma unroll
    for (int r = 0; r < 8; ++r)
        sOut[wave * 256 + (mh * 8 + r) * 16 + nl] = o[r];
    __syncthreads();
    {
        int row = tid >> 4, cq = tid & 15;
#pragma unroll
        for (int cs = 0; cs < 4; ++cs) {
            float t = sOut[cs * 256 + row * 16 + cq] + sOut[(cs + 4) * 256 + row * 16 + cq];
            Ctx[((size_t)b * SEQ + qt * 16 + row) * HIDDEN + h * HEAD_W + cs * 16 + cq] = t;
        }
    }
}

// ---------------------------------------------------------------------------
// LayerNorm over rows of 1024 (one block per row)
// ---------------------------------------------------------------------------
__global__ __launch_bounds__(256) void layernorm_rows(
    const float* __restrict__ X, const float* __restrict__ gamma,
    const float* __restrict__ beta, float* __restrict__ Y)
{
    __shared__ float red[2][8];
    const int tid = threadIdx.x, lane = tid & 31, wave = tid >> 5;
    const float* x = X + (size_t)blockIdx.x * HIDDEN;
    float v[4], s = 0.f, ss = 0.f;
#pragma unroll
    for (int i = 0; i < 4; ++i) {
        v[i] = x[tid + 256 * i];
        s += v[i]; ss += v[i] * v[i];
    }
    for (int off = 16; off; off >>= 1) { s += __shfl_xor(s, off); ss += __shfl_xor(ss, off); }
    if (lane == 0) { red[0][wave] = s; red[1][wave] = ss; }
    __syncthreads();
    if (wave == 0) {
        float a = (lane < 8) ? red[0][lane] : 0.f;
        float c = (lane < 8) ? red[1][lane] : 0.f;
        for (int off = 4; off; off >>= 1) { a += __shfl_xor(a, off); c += __shfl_xor(c, off); }
        if (lane == 0) { red[0][0] = a; red[1][0] = c; }
    }
    __syncthreads();
    float mean = red[0][0] * (1.0f / HIDDEN);
    float var  = red[1][0] * (1.0f / HIDDEN) - mean * mean;
    float inv  = rsqrtf(var + LN_EPS);
    float* y = Y + (size_t)blockIdx.x * HIDDEN;
#pragma unroll
    for (int i = 0; i < 4; ++i) {
        int c = tid + 256 * i;
        y[c] = (v[i] - mean) * inv * gamma[c] + beta[c];
    }
}

// ---------------------------------------------------------------------------
extern "C" void kernel_launch(void* const* d_in, const int* in_sizes, int n_in,
                              void* d_out, int out_size, void* d_ws, size_t ws_size,
                              hipStream_t stream) {
    (void)in_sizes; (void)n_in; (void)out_size; (void)ws_size;
    const float* q    = (const float*)d_in[0];
    const float* k    = (const float*)d_in[1];
    const float* v    = (const float*)d_in[2];
    const unsigned char* mask = (const unsigned char*)d_in[3];
    const float* Qw   = (const float*)d_in[4];
    const float* Qb   = (const float*)d_in[5];
    const float* Kw   = (const float*)d_in[6];
    const float* Kb   = (const float*)d_in[7];
    const float* Vw   = (const float*)d_in[8];
    const float* Vb   = (const float*)d_in[9];
    const float* Mw   = (const float*)d_in[10];
    const float* Mb   = (const float*)d_in[11];
    const float* ln_g = (const float*)d_in[12];
    const float* ln_b = (const float*)d_in[13];

    float* out_x    = (float*)d_out;                               // (B,S,D)
    float* out_attn = out_x + (size_t)ROWS * HIDDEN;               // (H*B,S,S)

    const size_t REG = (size_t)ROWS * HIDDEN;                      // 4M floats
    float* Qh  = (float*)d_ws;
    float* Kh  = Qh + REG;
    float* Vh  = Kh + REG;
    float* Ctx = Vh + REG;
    float* Tmp = Qh;   // Qh dead after attention -> reuse for projection output

    dim3 gGemm(HIDDEN / 64, ROWS / 128);   // (16, 32)

    gemm_xwt_bf16<<<gGemm, 128, 0, stream>>>(q, Qw, Qb, nullptr, Qh, ROWS, HIDDEN, HIDDEN);
    gemm_xwt_bf16<<<gGemm, 128, 0, stream>>>(k, Kw, Kb, nullptr, Kh, ROWS, HIDDEN, HIDDEN);
    gemm_xwt_bf16<<<gGemm, 128, 0, stream>>>(v, Vw, Vb, nullptr, Vh, ROWS, HIDDEN, HIDDEN);

    attn_fused<<<dim3(SEQ / 16, HEADS * BATCH), 256, 0, stream>>>(Qh, Kh, Vh, mask,
                                                                  out_attn, Ctx);

    gemm_xwt_bf16<<<gGemm, 128, 0, stream>>>(Ctx, Mw, Mb, q, Tmp, ROWS, HIDDEN, HIDDEN);
    layernorm_rows<<<ROWS, 256, 0, stream>>>(Tmp, ln_g, ln_b, out_x);
}
